// TransformerNLI_12850542150163
// MI455X (gfx1250) — compile-verified, matrix operand order checked by reading, not run on hardware
//
#include <hip/hip_runtime.h>
#include <math.h>

// ---------------- model dims ----------------
#define Ee   512
#define Hh   8
#define DK   64
#define FFd  2048
#define Ss   256
#define Bbb  128
#define TOK  (Ss * Bbb)            // 32768 tokens
#define LBL  3
#define D_THRESH 3.8918202981106265f   // log(2000/40 - 1) = log(49)

// ---------------- vector types ----------------
typedef __attribute__((ext_vector_type(16))) __bf16 v16bf;
typedef __attribute__((ext_vector_type(8)))  float  v8f;
typedef __attribute__((ext_vector_type(4)))  __bf16 v4bf;
typedef __attribute__((ext_vector_type(4)))  float  v4f;
typedef int v4i __attribute__((vector_size(16)));

union FragU { v16bf v; v4f f4[2]; };
union Bf8U  { v4f f; __bf16 h[8]; };

// ---------------- optional gfx1250 async global->LDS path ----------------
#if defined(__has_builtin)
#  if __has_builtin(__builtin_amdgcn_global_load_async_to_lds_b128) && \
      __has_builtin(__builtin_amdgcn_s_wait_asynccnt)
#    define HAVE_ASYNC_LDS 1
#  endif
#endif
#ifndef HAVE_ASYNC_LDS
#  define HAVE_ASYNC_LDS 0
#endif

// copy 16 bytes (8 bf16) global -> LDS, per lane
static __device__ __forceinline__ void cp16_g2s(const __bf16* g, __bf16* s) {
#if HAVE_ASYNC_LDS
  __builtin_amdgcn_global_load_async_to_lds_b128(
      (__attribute__((address_space(1))) v4i*)(v4i*)(g),
      (__attribute__((address_space(3))) v4i*)(v4i*)(s), 0, 0);
#else
  *(v4f*)s = *(const v4f*)g;
#endif
}
static __device__ __forceinline__ void async_join() {
#if HAVE_ASYNC_LDS
  __builtin_amdgcn_s_wait_asynccnt(0);
#endif
}

// Load a 16x32 bf16 operand fragment (A layout; B uses the same layout on the
// transposed matrix) from an LDS tile with row stride `stride` (elements).
static __device__ __forceinline__ v16bf lds_frag(const __bf16* base, int stride,
                                                 int row0, int kofs) {
  int lane = threadIdx.x & 31;
  const __bf16* p = base + (size_t)(row0 + (lane & 15)) * stride + kofs + ((lane & 16) ? 8 : 0);
  FragU u;
  u.f4[0] = *(const v4f*)(p);
  u.f4[1] = *(const v4f*)(p + 16);
  return u.v;
}

static __device__ __forceinline__ v8f wmma_bf16(v16bf a, v16bf b, v8f c) {
  return __builtin_amdgcn_wmma_f32_16x16x32_bf16(false, a, false, b, (short)0, c,
                                                 false, false);
}

static __device__ __forceinline__ unsigned fkey(float f) {
  unsigned u = __float_as_uint(f);
  return (u & 0x80000000u) ? ~u : (u | 0x80000000u);
}
static __device__ __forceinline__ float funkey(unsigned k) {
  unsigned u = (k & 0x80000000u) ? (k & 0x7fffffffu) : ~k;
  return __uint_as_float(u);
}

// ---------------- generic GEMM: C = [resid +] A(f32) x B(f32) [+bias] [relu] --------
// bf16 WMMA compute, fp32 accumulate, optional bf16 output.
// Requires M%128==0, N%64==0, K%64==0.
#define BM 128
#define BN 64
#define BK 64
#define AST 72   // LDS row stride (bf16 elems) for A tile  [BM][AST]
#define BST 72   // LDS row stride (bf16 elems) for B^T tile [BN][BST]

template<bool OUTBF16, bool BIAS, bool RESID, bool RELU>
__global__ __launch_bounds__(256)
void gemm_k(const float* __restrict__ A, int lda,
            const float* __restrict__ B, int ldb,
            const float* __restrict__ bias,
            const float* resid,
            void* Cv, int ldc,
            int M, int N, int K)
{
  __shared__ __bf16 As[BM * AST];
  __shared__ __bf16 Bs[BN * BST];   // transposed: [n][k]

  int tid = threadIdx.x;
  int m0 = blockIdx.y * BM;
  int n0 = blockIdx.x * BN;
  int w = tid >> 5, lane = tid & 31;
  int wm = (w & 3) * 32;   // 4 waves along M
  int wn = (w >> 2) * 32;  // 2 waves along N

  const v8f vzero = {0.f, 0.f, 0.f, 0.f, 0.f, 0.f, 0.f, 0.f};
  v8f acc[2][2];
  acc[0][0] = vzero; acc[0][1] = vzero; acc[1][0] = vzero; acc[1][1] = vzero;

  for (int k0 = 0; k0 < K; k0 += BK) {
    // stage A: 128x64 f32 -> bf16 (2048 v4f, 8 per thread)
    #pragma unroll
    for (int i = 0; i < 8; i++) {
      int q = tid + i * 256;
      int r = q >> 4, kq = (q & 15) * 4;
      v4f v = *(const v4f*)(A + (size_t)(m0 + r) * lda + k0 + kq);
      v4bf hb = { (__bf16)v.x, (__bf16)v.y, (__bf16)v.z, (__bf16)v.w };
      *(v4bf*)&As[r * AST + kq] = hb;
    }
    // stage B transposed: 64x64 -> Bs[n][k] (1024 v4f, 4 per thread)
    #pragma unroll
    for (int i = 0; i < 4; i++) {
      int q = tid + i * 256;
      int kr = q >> 4, nq = (q & 15) * 4;
      v4f v = *(const v4f*)(B + (size_t)(k0 + kr) * ldb + n0 + nq);
      Bs[(nq + 0) * BST + kr] = (__bf16)v.x;
      Bs[(nq + 1) * BST + kr] = (__bf16)v.y;
      Bs[(nq + 2) * BST + kr] = (__bf16)v.z;
      Bs[(nq + 3) * BST + kr] = (__bf16)v.w;
    }
    // prefetch next K tiles into cache (global_prefetch_b8)
    if (k0 + BK < K) {
      __builtin_prefetch(A + (size_t)(m0 + (tid >> 1)) * lda + k0 + BK, 0, 1);
      __builtin_prefetch(B + (size_t)(k0 + BK + (tid & 63)) * ldb + n0, 0, 1);
    }
    __syncthreads();

    #pragma unroll
    for (int kk = 0; kk < BK; kk += 32) {
      v16bf a0 = lds_frag(As, AST, wm + 0, kk);
      v16bf a1 = lds_frag(As, AST, wm + 16, kk);
      v16bf b0 = lds_frag(Bs, BST, wn + 0, kk);
      v16bf b1 = lds_frag(Bs, BST, wn + 16, kk);
      acc[0][0] = wmma_bf16(a0, b0, acc[0][0]);
      acc[0][1] = wmma_bf16(a0, b1, acc[0][1]);
      acc[1][0] = wmma_bf16(a1, b0, acc[1][0]);
      acc[1][1] = wmma_bf16(a1, b1, acc[1][1]);
    }
    __syncthreads();
  }

  int rbase = (lane & 16) ? 8 : 0;
  int col = lane & 15;
  #pragma unroll
  for (int ti = 0; ti < 2; ti++) {
    #pragma unroll
    for (int tj = 0; tj < 2; tj++) {
      #pragma unroll
      for (int e = 0; e < 8; e++) {
        int r = m0 + wm + ti * 16 + rbase + e;
        int c = n0 + wn + tj * 16 + col;
        size_t idx = (size_t)r * ldc + c;
        float val = acc[ti][tj][e];
        if constexpr (BIAS)  val += bias[c];
        if constexpr (RESID) val += resid[idx];
        if constexpr (RELU)  val = fmaxf(val, 0.f);
        if constexpr (OUTBF16) ((__bf16*)Cv)[idx] = (__bf16)val;
        else                   ((float*)Cv)[idx] = val;
      }
    }
  }
}

// ---------------- attention (threshold softmax), one block per (b,h,64-query tile) ----
#define QT 64

__global__ __launch_bounds__(256)
void attention_k(const __bf16* __restrict__ qkv, float* __restrict__ out)
{
  __shared__ __bf16 Qt[QT * 72];      // [q][dk]
  __shared__ __bf16 Kt[Ss * 72];      // [key][dk]
  __shared__ __bf16 Vt[DK * 264];     // [dk][key]  (transposed)
  __shared__ __bf16 Wt[QT * 264];     // [q][key]   softmax weights
  __shared__ unsigned rowMaxU[QT];
  __shared__ float rowSum[QT];

  int blk = blockIdx.x;               // B*H*(S/QT) = 4096
  int b = blk >> 5;
  int rem = blk & 31;
  int h = rem >> 2;
  int q0 = (rem & 3) * QT;
  int tid = threadIdx.x;

  if (tid < QT) { rowMaxU[tid] = 0u; rowSum[tid] = 0.f; }

  const int ld = 3 * Ee;              // 1536 (bf16 elems)
  const __bf16* qbase = qkv + h * DK;
  const __bf16* kbase = qkv + Ee + h * DK;
  const __bf16* vbase = qkv + 2 * Ee + h * DK;

  // load Q: 64 rows x 64 dk bf16 = 512 16B chunks (2 per thread), async to LDS
  #pragma unroll
  for (int i = 0; i < 2; i++) {
    int q = tid + i * 256;
    int r = q >> 3, c8 = (q & 7) * 8;
    int token = (q0 + r) * Bbb + b;
    cp16_g2s(qbase + (size_t)token * ld + c8, &Qt[r * 72 + c8]);
  }
  // load K: 256 rows x 64 dk = 2048 chunks (8 per thread), async to LDS
  #pragma unroll
  for (int i = 0; i < 8; i++) {
    int q = tid + i * 256;
    int r = q >> 3, c8 = (q & 7) * 8;
    int token = r * Bbb + b;
    cp16_g2s(kbase + (size_t)token * ld + c8, &Kt[r * 72 + c8]);
  }
  // load V transposed (needs element shuffle -> synchronous)
  #pragma unroll
  for (int i = 0; i < 8; i++) {
    int q = tid + i * 256;
    int key = q >> 3, c8 = (q & 7) * 8;
    int token = key * Bbb + b;
    Bf8U u;
    u.f = *(const v4f*)(vbase + (size_t)token * ld + c8);
    #pragma unroll
    for (int j = 0; j < 8; j++) Vt[(c8 + j) * 264 + key] = u.h[j];
  }
  async_join();
  __syncthreads();

  int w = tid >> 5, lane = tid & 31;
  int qb = (w & 3) * 16;        // 4 q-blocks
  int kh = (w >> 2) * 8;        // waves 0-3: key blocks 0..7, waves 4-7: 8..15
  int rbase = (lane & 16) ? 8 : 0;
  int col = lane & 15;
  const v8f vzero = {0.f, 0.f, 0.f, 0.f, 0.f, 0.f, 0.f, 0.f};
  const float qscale = 0.125f;  // dk^-0.5 applied to scores

  v16bf aq0 = lds_frag(Qt, 72, qb, 0);
  v16bf aq1 = lds_frag(Qt, 72, qb, 32);

  // pass 1: row max
  float lmax[8];
  #pragma unroll
  for (int e = 0; e < 8; e++) lmax[e] = -3.0e38f;
  for (int kb = 0; kb < 8; kb++) {
    int key0 = (kh + kb) * 16;
    v8f acc = vzero;
    acc = wmma_bf16(aq0, lds_frag(Kt, 72, key0, 0),  acc);
    acc = wmma_bf16(aq1, lds_frag(Kt, 72, key0, 32), acc);
    #pragma unroll
    for (int e = 0; e < 8; e++) lmax[e] = fmaxf(lmax[e], acc[e] * qscale);
  }
  #pragma unroll
  for (int e = 0; e < 8; e++)
    atomicMax(&rowMaxU[qb + rbase + e], fkey(lmax[e]));
  __syncthreads();

  // pass 2: recompute, threshold+exp, store bf16 weights, row sums
  float lsum[8] = {0.f, 0.f, 0.f, 0.f, 0.f, 0.f, 0.f, 0.f};
  for (int kb = 0; kb < 8; kb++) {
    int key0 = (kh + kb) * 16;
    v8f acc = vzero;
    acc = wmma_bf16(aq0, lds_frag(Kt, 72, key0, 0),  acc);
    acc = wmma_bf16(aq1, lds_frag(Kt, 72, key0, 32), acc);
    #pragma unroll
    for (int e = 0; e < 8; e++) {
      int r = qb + rbase + e;
      float mx = funkey(rowMaxU[r]);
      float sc = acc[e] * qscale;
      float wgt = (sc >= mx - D_THRESH) ? __expf(sc - mx) : 0.f;
      lsum[e] += wgt;
      Wt[r * 264 + key0 + col] = (__bf16)wgt;
    }
  }
  #pragma unroll
  for (int e = 0; e < 8; e++)
    atomicAdd(&rowSum[qb + rbase + e], lsum[e]);
  __syncthreads();

  // pass 3: o = W @ V, normalize by rowSum
  for (int t = w; t < 16; t += 8) {
    int qq = (t & 3) * 16, nb = (t >> 2) * 16;
    v8f acc = vzero;
    for (int ks = 0; ks < Ss; ks += 32)
      acc = wmma_bf16(lds_frag(Wt, 264, qq, ks), lds_frag(Vt, 264, nb, ks), acc);
    #pragma unroll
    for (int e = 0; e < 8; e++) {
      int r = qq + rbase + e;
      int token = (q0 + r) * Bbb + b;
      out[(size_t)token * Ee + h * DK + nb + col] = acc[e] / rowSum[r];
    }
  }
}

// ---------------- small elementwise kernels ----------------
__global__ __launch_bounds__(256)
void peadd_k(const float* __restrict__ x, float* __restrict__ y)
{
  int i = blockIdx.x * 256 + threadIdx.x;      // over S*B*E
  int s = i / (Bbb * Ee);
  int e = i & (Ee - 1);
  int j = e >> 1;
  float d = __expf((float)(2 * j) * (-9.210340371976184f / (float)Ee));
  float ang = (float)s * d;
  float pe = (e & 1) ? __cosf(ang) : __sinf(ang);
  y[i] = x[i] + pe;
}

__global__ __launch_bounds__(256)
void layernorm_k(const float* __restrict__ x, const float* __restrict__ g,
                 const float* __restrict__ b, float* __restrict__ y)
{
  __shared__ float red[256];
  size_t base = (size_t)blockIdx.x * Ee;
  int t = threadIdx.x;
  float v0 = x[base + t], v1 = x[base + t + 256];
  red[t] = v0 + v1;
  __syncthreads();
  for (int o = 128; o > 0; o >>= 1) { if (t < o) red[t] += red[t + o]; __syncthreads(); }
  float mean = red[0] * (1.0f / Ee);
  __syncthreads();
  float d0 = v0 - mean, d1 = v1 - mean;
  red[t] = d0 * d0 + d1 * d1;
  __syncthreads();
  for (int o = 128; o > 0; o >>= 1) { if (t < o) red[t] += red[t + o]; __syncthreads(); }
  float inv = rsqrtf(red[0] * (1.0f / Ee) + 1e-5f);
  y[base + t]       = d0 * inv * g[t]       + b[t];
  y[base + t + 256] = d1 * inv * g[t + 256] + b[t + 256];
}

__global__ __launch_bounds__(256)
void pool_k(const float* __restrict__ x, float* __restrict__ p)
{
  int i = blockIdx.x * 256 + threadIdx.x;      // B*E
  int bI = i / Ee, e = i & (Ee - 1);
  float s = 0.f;
  for (int ss = 0; ss < Ss; ss++) s += x[(size_t)(ss * Bbb + bI) * Ee + e];
  p[i] = s * 0.0625f;                           // / sqrt(256)
}

__global__ __launch_bounds__(256)
void cat2_k(const float* __restrict__ a, const float* __restrict__ bsrc,
            float* __restrict__ o)
{
  int i = blockIdx.x * 256 + threadIdx.x;      // rows * 2E
  int t = i / (2 * Ee), c = i - t * 2 * Ee;
  o[i] = (c < Ee) ? a[(size_t)t * Ee + c] : bsrc[(size_t)t * Ee + (c - Ee)];
}

__global__ void fc4_k(const float* __restrict__ t, const float* __restrict__ wmat,
                      const float* __restrict__ bias, float* __restrict__ out)
{
  int i = blockIdx.x * 64 + threadIdx.x;       // B*LBL = 384
  if (i >= Bbb * LBL) return;
  int bI = i / LBL, c = i - bI * LBL;
  float s = bias[c];
  for (int k = 0; k < Ee; k++) s += t[bI * Ee + k] * wmat[k * LBL + c];
  out[i] = s;
}

// ---------------- host-side launch helpers ----------------
static void gemm_f32out(hipStream_t st, const float* A, int lda, const float* B, int ldb,
                        const float* bias, const float* resid, float* C, int ldc,
                        int M, int N, int K, int relu)
{
  dim3 g(N / BN, M / BM);
  if (resid)
    gemm_k<false, true, true, false><<<g, 256, 0, st>>>(A, lda, B, ldb, bias, resid, C, ldc, M, N, K);
  else if (relu)
    gemm_k<false, true, false, true><<<g, 256, 0, st>>>(A, lda, B, ldb, bias, nullptr, C, ldc, M, N, K);
  else
    gemm_k<false, true, false, false><<<g, 256, 0, st>>>(A, lda, B, ldb, bias, nullptr, C, ldc, M, N, K);
}
static void gemm_bf16out(hipStream_t st, const float* A, int lda, const float* B, int ldb,
                         const float* bias, __bf16* C, int ldc, int M, int N, int K)
{
  dim3 g(N / BN, M / BM);
  gemm_k<true, true, false, false><<<g, 256, 0, st>>>(A, lda, B, ldb, bias, nullptr, C, ldc, M, N, K);
}
static void lnorm(hipStream_t st, const float* x, const float* g, const float* b, float* y)
{
  layernorm_k<<<TOK, 256, 0, st>>>(x, g, b, y);
}
static void attn(hipStream_t st, const __bf16* qkv, float* out)
{
  attention_k<<<Bbb * Hh * (Ss / QT), 256, 0, st>>>(qkv, out);
}
static void mha_self(hipStream_t st, float* X, const float* h,
                     const float* Wqkv, const float* bqkv,
                     const float* Wo, const float* bo, __bf16* QKVb, float* Tb)
{
  gemm_bf16out(st, h, Ee, Wqkv, 3 * Ee, bqkv, QKVb, 3 * Ee, TOK, 3 * Ee, Ee);
  attn(st, QKVb, Tb);
  gemm_f32out(st, Tb, Ee, Wo, Ee, bo, X, X, Ee, TOK, Ee, Ee, 0);
}
static void mha_cross(hipStream_t st, float* X, const float* hq, const float* hkv,
                      const float* Wqkv, const float* bqkv,
                      const float* Wo, const float* bo, __bf16* QKVb, float* Tb)
{
  gemm_bf16out(st, hq, Ee, Wqkv, 3 * Ee, bqkv, QKVb, 3 * Ee, TOK, Ee, Ee);
  gemm_bf16out(st, hkv, Ee, Wqkv + Ee, 3 * Ee, bqkv + Ee, QKVb + Ee, 3 * Ee, TOK, 2 * Ee, Ee);
  attn(st, QKVb, Tb);
  gemm_f32out(st, Tb, Ee, Wo, Ee, bo, X, X, Ee, TOK, Ee, Ee, 0);
}
static void ffn_block(hipStream_t st, float* X, const float* h,
                      const float* W1, const float* b1,
                      const float* W2, const float* b2, float* mid)
{
  gemm_f32out(st, h, Ee, W1, FFd, b1, nullptr, mid, FFd, TOK, FFd, Ee, 1);
  gemm_f32out(st, mid, FFd, W2, Ee, b2, X, X, Ee, TOK, Ee, FFd, 0);
}

extern "C" void kernel_launch(void* const* d_in, const int* in_sizes, int n_in,
                              void* d_out, int out_size, void* d_ws, size_t ws_size,
                              hipStream_t stream)
{
  (void)in_sizes; (void)n_in; (void)out_size; (void)ws_size;

  const float* x1_in     = (const float*)d_in[0];
  const float* x2_in     = (const float*)d_in[1];
  const float* enc_ln1_g = (const float*)d_in[2];
  const float* enc_ln1_b = (const float*)d_in[3];
  const float* enc_Wqkv  = (const float*)d_in[4];
  const float* enc_bqkv  = (const float*)d_in[5];
  const float* enc_Wo    = (const float*)d_in[6];
  const float* enc_bo    = (const float*)d_in[7];
  const float* enc_ln2_g = (const float*)d_in[8];
  const float* enc_ln2_b = (const float*)d_in[9];
  const float* enc_W1    = (const float*)d_in[10];
  const float* enc_b1    = (const float*)d_in[11];
  const float* enc_W2    = (const float*)d_in[12];
  const float* enc_b2    = (const float*)d_in[13];
  const float* isa_ln_g  = (const float*)d_in[14];
  const float* isa_ln_b  = (const float*)d_in[15];
  const float* isa_Wqkv  = (const float*)d_in[16];
  const float* isa_bqkv  = (const float*)d_in[17];
  const float* isa_Wo    = (const float*)d_in[18];
  const float* isa_bo    = (const float*)d_in[19];
  const float* iea_ln_g  = (const float*)d_in[20];
  const float* iea_ln_b  = (const float*)d_in[21];
  const float* iea_Wqkv  = (const float*)d_in[22];
  const float* iea_bqkv  = (const float*)d_in[23];
  const float* iea_Wo    = (const float*)d_in[24];
  const float* iea_bo    = (const float*)d_in[25];
  const float* iff_ln_g  = (const float*)d_in[26];
  const float* iff_ln_b  = (const float*)d_in[27];
  const float* int_W1    = (const float*)d_in[28];
  const float* int_b1    = (const float*)d_in[29];
  const float* int_W2    = (const float*)d_in[30];
  const float* int_b2    = (const float*)d_in[31];
  const float* fc1_w     = (const float*)d_in[32];
  const float* fc1_b     = (const float*)d_in[33];
  const float* fc2_w     = (const float*)d_in[34];
  const float* fc2_b     = (const float*)d_in[35];
  const float* fc3_w     = (const float*)d_in[36];
  const float* fc3_b     = (const float*)d_in[37];
  const float* fc4_w     = (const float*)d_in[38];
  const float* fc4_b     = (const float*)d_in[39];

  // workspace layout (float units)
  const size_t SBEsz = (size_t)TOK * Ee;          // 16.78M floats
  float* ws   = (float*)d_ws;
  float* X1   = ws;
  float* X2   = X1 + SBEsz;
  float* E1b  = X2 + SBEsz;
  float* E2b  = E1b + SBEsz;
  float* H1b  = E2b + SBEsz;
  float* H2b  = H1b + SBEsz;
  float* QREG = H2b + SBEsz;                       // 2*SBE floats: qkv(bf16, 1.5*SBE fl) / concat(2*SBE fl)
  __bf16* QKVb = (__bf16*)QREG;
  float*  CATb = QREG;
  float* T3b  = QREG + 2 * SBEsz;                  // TOK*FF scratch (4*SBE floats)
  float* SM   = T3b + (size_t)TOK * FFd;
  float* p1   = SM;
  float* p2   = p1 + (size_t)Bbb * Ee;
  float* pcat = p2 + (size_t)Bbb * Ee;
  float* tb   = pcat + (size_t)2 * Bbb * Ee;

  const int PEB = (TOK * Ee) / 256;
  peadd_k<<<PEB, 256, 0, stream>>>(x1_in, X1);
  peadd_k<<<PEB, 256, 0, stream>>>(x2_in, X2);

  // Siamese encoders (shared weights)
  for (int i = 0; i < 3; i++) {
    float* Xs[2] = {X1, X2};
    for (int s = 0; s < 2; s++) {
      float* X = Xs[s];
      lnorm(stream, X, enc_ln1_g + i * Ee, enc_ln1_b + i * Ee, H1b);
      mha_self(stream, X, H1b, enc_Wqkv + (size_t)i * Ee * 3 * Ee, enc_bqkv + i * 3 * Ee,
               enc_Wo + (size_t)i * Ee * Ee, enc_bo + i * Ee, QKVb, T3b);
      lnorm(stream, X, enc_ln2_g + i * Ee, enc_ln2_b + i * Ee, H1b);
      ffn_block(stream, X, H1b, enc_W1 + (size_t)i * Ee * FFd, enc_b1 + i * FFd,
                enc_W2 + (size_t)i * FFd * Ee, enc_b2 + i * Ee, T3b);
    }
  }
  (void)hipMemcpyAsync(E1b, X1, SBEsz * sizeof(float), hipMemcpyDeviceToDevice, stream);
  (void)hipMemcpyAsync(E2b, X2, SBEsz * sizeof(float), hipMemcpyDeviceToDevice, stream);

  // interaction stack
  for (int i = 0; i < 3; i++) {
    // self-attention (both streams)
    lnorm(stream, X1, isa_ln_g + i * Ee, isa_ln_b + i * Ee, H1b);
    mha_self(stream, X1, H1b, isa_Wqkv + (size_t)i * Ee * 3 * Ee, isa_bqkv + i * 3 * Ee,
             isa_Wo + (size_t)i * Ee * Ee, isa_bo + i * Ee, QKVb, T3b);
    lnorm(stream, X2, isa_ln_g + i * Ee, isa_ln_b + i * Ee, H1b);
    mha_self(stream, X2, H1b, isa_Wqkv + (size_t)i * Ee * 3 * Ee, isa_bqkv + i * 3 * Ee,
             isa_Wo + (size_t)i * Ee * Ee, isa_bo + i * Ee, QKVb, T3b);
    // cross-attention (h1/h2 from pre-update x)
    lnorm(stream, X1, iea_ln_g + i * Ee, iea_ln_b + i * Ee, H1b);
    lnorm(stream, X2, iea_ln_g + i * Ee, iea_ln_b + i * Ee, H2b);
    mha_cross(stream, X1, H1b, H2b, iea_Wqkv + (size_t)i * Ee * 3 * Ee,
              iea_bqkv + i * 3 * Ee, iea_Wo + (size_t)i * Ee * Ee, iea_bo + i * Ee,
              QKVb, T3b);
    mha_cross(stream, X2, H2b, H1b, iea_Wqkv + (size_t)i * Ee * 3 * Ee,
              iea_bqkv + i * 3 * Ee, iea_Wo + (size_t)i * Ee * Ee, iea_bo + i * Ee,
              QKVb, T3b);
    // FFN (both streams)
    lnorm(stream, X1, iff_ln_g + i * Ee, iff_ln_b + i * Ee, H1b);
    ffn_block(stream, X1, H1b, int_W1 + (size_t)i * Ee * FFd, int_b1 + i * FFd,
              int_W2 + (size_t)i * FFd * Ee, int_b2 + i * Ee, T3b);
    lnorm(stream, X2, iff_ln_g + i * Ee, iff_ln_b + i * Ee, H1b);
    ffn_block(stream, X2, H1b, int_W1 + (size_t)i * Ee * FFd, int_b1 + i * FFd,
              int_W2 + (size_t)i * FFd * Ee, int_b2 + i * Ee, T3b);
  }

  // classifier
  const int CATB = (TOK * 2 * Ee) / 256;
  cat2_k<<<CATB, 256, 0, stream>>>(E1b, X1, CATb);
  gemm_f32out(stream, CATb, 2 * Ee, fc1_w, Ee, fc1_b, nullptr, H1b, Ee, TOK, Ee, 2 * Ee, 1);
  gemm_f32out(stream, H1b, Ee, fc2_w, Ee, fc2_b, nullptr, T3b, Ee, TOK, Ee, Ee, 0);
  pool_k<<<(Bbb * Ee) / 256, 256, 0, stream>>>(T3b, p1);

  cat2_k<<<CATB, 256, 0, stream>>>(E2b, X2, CATb);
  gemm_f32out(stream, CATb, 2 * Ee, fc1_w, Ee, fc1_b, nullptr, H1b, Ee, TOK, Ee, 2 * Ee, 1);
  gemm_f32out(stream, H1b, Ee, fc2_w, Ee, fc2_b, nullptr, T3b, Ee, TOK, Ee, Ee, 0);
  pool_k<<<(Bbb * Ee) / 256, 256, 0, stream>>>(T3b, p2);

  cat2_k<<<(Bbb * 2 * Ee) / 256, 256, 0, stream>>>(p1, p2, pcat);
  gemm_f32out(stream, pcat, 2 * Ee, fc3_w, Ee, fc3_b, nullptr, tb, Ee, Bbb, Ee, 2 * Ee, 1);
  fc4_k<<<(Bbb * LBL + 63) / 64, 64, 0, stream>>>(tb, fc4_w, fc4_b, (float*)d_out);
}
